// DiagonalSSM_86732569575473
// MI455X (gfx1250) — compile-verified, matrix operand order checked by reading, not run on hardware
//
#include <hip/hip_runtime.h>
#include <hip/hip_bf16.h>

// ---------------- problem constants ----------------
#define BB   4
#define TT   2048
#define DD   1024
#define HH   4096
#define EPSV 1e-6f
#define NTOK (BB * TT)          // 8192
#define NC   16                 // scan chunks
#define CL   (TT / NC)          // 128

// ---------------- vector types ----------------
typedef __bf16        v16bf __attribute__((ext_vector_type(16)));
typedef float         v8f   __attribute__((ext_vector_type(8)));
typedef unsigned int  u32x4 __attribute__((ext_vector_type(4)));
typedef int           i32x4 __attribute__((ext_vector_type(4)));
typedef int           i32x8 __attribute__((ext_vector_type(8)));

static constexpr int SA = 34;   // LDS row stride (halves): 32 + 1 dword pad (TDM pad matches)
static constexpr int SB = 34;

__device__ __forceinline__ v8f wmma_bf16(v16bf a, v16bf b, v8f c) {
  return __builtin_amdgcn_wmma_f32_16x16x32_bf16(false, a, false, b, (short)0, c, false, false);
}

// Build a 16x32 bf16 fragment from a padded LDS tile.
// lane l: row = l%16 (caller folds in subtile offset), kbase = (l/16)*8,
// VGPR v holds halves k = kbase + (v%4)*2 + (v/4)*16 (k-contiguous pairs -> ds b32 loads)
__device__ __forceinline__ v16bf frag_from_lds(const unsigned short* base, int row, int stride, int lane) {
  v16bf f;
  unsigned int* fp = (unsigned int*)&f;
  const int klane = (lane >> 4) * 8;
#pragma unroll
  for (int v = 0; v < 8; ++v) {
    const int k = klane + (v & 3) * 2 + (v >> 2) * 16;
    fp[v] = *(const unsigned int*)&base[row * stride + k];
  }
  return f;
}

// ---------------- Tensor Data Mover: 2D bf16 tile -> LDS ----------------
// Loads a tile_m x tile_k (rows x contiguous-halves) bf16 tile into LDS with one
// pad dword appended per 16 dwords (one 32-half row) => LDS row stride 34 halves.
__device__ __forceinline__ void tdm_load_tile_bf16(const void* gsrc, unsigned lds_off,
                                                   unsigned tensor_d0, unsigned tensor_d1,
                                                   unsigned stride0_elems,
                                                   unsigned tile_k, unsigned tile_m) {
  const unsigned long long ga = (unsigned long long)gsrc;
  u32x4 g0;
  g0[0] = 1u;                                       // count=1, no gather, user desc
  g0[1] = lds_off;                                  // lds_addr (bytes)
  g0[2] = (unsigned)(ga & 0xffffffffu);             // global_addr[31:0]
  g0[3] = (unsigned)((ga >> 32) & 0x01ffffffu)      // global_addr[56:32]
        | (2u << 30);                               // type = 2 ("image")
  i32x8 g1;
  g1[0] = (int)((1u << 16)                          // data_size = 1 (2 bytes)
              | (1u << 20)                          // pad_enable
              | (3u << 22)                          // pad_interval: 16 dwords
              | (0u << 25));                        // pad_amount: 1 dword
  g1[1] = (int)((tensor_d0 & 0xffffu) << 16);                       // dim0[15:0]
  g1[2] = (int)((tensor_d0 >> 16) | ((tensor_d1 & 0xffffu) << 16)); // dim0[31:16], dim1[15:0]
  g1[3] = (int)((tensor_d1 >> 16) | (tile_k << 16));                // dim1[31:16], tile_dim0
  g1[4] = (int)(tile_m & 0xffffu);                                  // tile_dim1 (tile_dim2=0)
  g1[5] = (int)stride0_elems;                                       // dim0_stride[31:0]
  g1[6] = 0;
  g1[7] = 0;
  const i32x4 z4 = {0, 0, 0, 0};
#if __has_include(<hip/amd_detail/amd_gfx1250_TDM.h>)
  const i32x8 z8 = {0, 0, 0, 0, 0, 0, 0, 0};
  __builtin_amdgcn_tensor_load_to_lds(g0, g1, z4, z4, z8, 0);
#else
  __builtin_amdgcn_tensor_load_to_lds(g0, g1, z4, z4, 0);
#endif
}

// ---------------- small utility kernels ----------------
__global__ void f32_to_bf16_kernel(const float* __restrict__ src, __bf16* __restrict__ dst, int n) {
  int i = blockIdx.x * blockDim.x + threadIdx.x;
  if (i < n) dst[i] = (__bf16)src[i];
}

template <bool BF16OUT>
__global__ void __launch_bounds__(256)
rmsnorm_kernel(const float* __restrict__ x, const float* __restrict__ w, void* __restrict__ out) {
  const int tok = blockIdx.x;
  const float* xr = x + (size_t)tok * DD;
  float ss = 0.f;
#pragma unroll
  for (int i = threadIdx.x; i < DD; i += 256) {
    float v = xr[i];
    ss += v * v;
  }
#pragma unroll
  for (int off = 16; off > 0; off >>= 1) ss += __shfl_xor(ss, off, 32);
  __shared__ float sred[8];
  if ((threadIdx.x & 31) == 0) sred[threadIdx.x >> 5] = ss;
  __syncthreads();
  float tot = 0.f;
#pragma unroll
  for (int j = 0; j < 8; ++j) tot += sred[j];
  const float inv = rsqrtf(tot * (1.0f / DD) + EPSV);
#pragma unroll
  for (int i = threadIdx.x; i < DD; i += 256) {
    float v = xr[i] * inv * w[i];
    if (BF16OUT) ((__bf16*)out)[(size_t)tok * DD + i] = (__bf16)v;
    else         ((float*)out)[(size_t)tok * DD + i]  = v;
  }
}

// Depthwise conv (K=3, pad=1) + chunk-local diagonal scan. gid = ((c*B)+b)*D + d
__global__ void __launch_bounds__(256)
conv_ssm_local_kernel(const float* __restrict__ x, const float* __restrict__ y,
                      const float* __restrict__ conv_w, const float* __restrict__ conv_b,
                      const float* __restrict__ a, const float* __restrict__ bvec,
                      float* __restrict__ xr, float* __restrict__ carry) {
  const int gid = blockIdx.x * blockDim.x + threadIdx.x;
  const int d = gid % DD;
  const int b = (gid / DD) % BB;
  const int c = gid / (BB * DD);
  const int t0 = c * CL;

  const float w0 = conv_w[d * 3 + 0], w1 = conv_w[d * 3 + 1], w2 = conv_w[d * 3 + 2];
  const float cb = conv_b[d];
  const float s  = tanhf(a[d]);
  const float bd = bvec[d];
  const size_t base = (size_t)b * TT * DD + d;

  float ym = (t0 > 0) ? y[base + (size_t)(t0 - 1) * DD] : 0.f;
  float yc = y[base + (size_t)t0 * DD];
  float h = 0.f;
  for (int tl = 0; tl < CL; ++tl) {
    const int t = t0 + tl;
    const float yp = (t + 1 < TT) ? y[base + (size_t)(t + 1) * DD] : 0.f;
    const float conv = cb + w0 * ym + w1 * yc + w2 * yp;
    h = s * h + bd * yc;  // chunk-local scan (h_in = 0; fixed up later)
    xr[base + (size_t)t * DD] = x[base + (size_t)t * DD] + conv + h;
    ym = yc;
    yc = yp;
  }
  carry[((size_t)b * DD + d) * NC + c] = h;
}

__global__ void __launch_bounds__(256)
scan_carries_kernel(const float* __restrict__ a, const float* __restrict__ carry,
                    float* __restrict__ prefix) {
  const int gid = blockIdx.x * blockDim.x + threadIdx.x;
  const int d = gid % DD;
  const float s = tanhf(a[d]);
  float p = s;  // s^128 via 7 squarings
#pragma unroll
  for (int i = 0; i < 7; ++i) p *= p;
  float run = 0.f;
#pragma unroll
  for (int c = 0; c < NC; ++c) {
    prefix[(size_t)gid * NC + c] = run;
    run = p * run + carry[(size_t)gid * NC + c];
  }
}

__global__ void __launch_bounds__(256)
ssm_fixup_kernel(const float* __restrict__ a, const float* __restrict__ prefix,
                 float* __restrict__ xr) {
  const int gid = blockIdx.x * blockDim.x + threadIdx.x;
  const int d = gid % DD;
  const int b = (gid / DD) % BB;
  const int c = gid / (BB * DD);
  const float pin = prefix[((size_t)b * DD + d) * NC + c];
  if (pin == 0.f) return;
  const float s = tanhf(a[d]);
  const size_t base = (size_t)b * TT * DD + d + (size_t)c * CL * DD;
  float f = s;
  for (int tl = 0; tl < CL; ++tl) {
    xr[base + (size_t)tl * DD] += f * pin;
    f *= s;
  }
}

// ---------------- bf16 WMMA GEMM kernels (double-buffered pipeline) ----------------
// B tile staging: 32(k) x 64(n) bf16, one b128 global load per thread, transposed on
// the LDS-store side into Bt[n][k] so B fragments use the same pattern as A.
__device__ __forceinline__ u32x4 load_b_regs(const __bf16* __restrict__ src, int ld,
                                             int k0, int n0) {
  const int k = threadIdx.x >> 3;       // 0..31
  const int q = threadIdx.x & 7;        // covers n = q*8 .. q*8+7
  return *(const u32x4*)(src + (size_t)(k0 + k) * ld + n0 + q * 8);
}

__device__ __forceinline__ void store_b_lds(u32x4 v, unsigned short* Bt) {
  const int k = threadIdx.x >> 3;
  const int q = threadIdx.x & 7;
#pragma unroll
  for (int e = 0; e < 4; ++e) {
    Bt[(q * 8 + 2 * e + 0) * SB + k] = (unsigned short)(v[e] & 0xffffu);
    Bt[(q * 8 + 2 * e + 1) * SB + k] = (unsigned short)(v[e] >> 16);
  }
}

// GEMM1: h = silu(z@Wg + bg) * (z@Wu + bu). M=8192, N=4096, K=1024.
// Block tile 128x64, BK=32, 8 waves as 4(m) x 2(n) groups, wave owns 2x2 subtiles.
// Double-buffered: TDM(A) + b128(B) for tile it+1 overlap WMMA compute of tile it.
__global__ void __launch_bounds__(256)
gemm_gate_up_kernel(const __bf16* __restrict__ z, const __bf16* __restrict__ wg,
                    const __bf16* __restrict__ wu, const float* __restrict__ bg,
                    const float* __restrict__ bu, __bf16* __restrict__ hbuf) {
  const int m0 = blockIdx.x * 128;
  const int n0 = blockIdx.y * 64;
  const int lane = threadIdx.x & 31;
  const int wm = (threadIdx.x >> 5) >> 1;  // 0..3
  const int wn = (threadIdx.x >> 5) & 1;   // 0..1

  __shared__ unsigned short As[2][128 * SA];
  __shared__ unsigned short Bgs[2][64 * SB];
  __shared__ unsigned short Bus[2][64 * SB];

  v8f accg[2][2] = {};   // [sm][sn]
  v8f accu[2][2] = {};

  constexpr int NIT = DD / 32;

  // prologue: stage tile 0 into bank 0
  if (threadIdx.x < 32)
    tdm_load_tile_bf16(z + (size_t)m0 * DD, (unsigned)(size_t)(void*)As[0],
                       DD, NTOK, DD, 32, 128);
  store_b_lds(load_b_regs(wg, HH, 0, n0), Bgs[0]);
  store_b_lds(load_b_regs(wu, HH, 0, n0), Bus[0]);
  if (threadIdx.x < 32) __builtin_amdgcn_s_wait_tensorcnt(0);

  for (int it = 0; it < NIT; ++it) {
    const int buf = it & 1;
    __syncthreads();                    // bank 'buf' is ready (dscnt + tensorcnt retired)

    // kick off tile it+1 into the other bank (overlaps this tile's WMMAs)
    u32x4 ng, nu;
    const bool more = (it + 1) < NIT;
    if (more) {
      ng = load_b_regs(wg, HH, (it + 1) * 32, n0);
      nu = load_b_regs(wu, HH, (it + 1) * 32, n0);
      if (threadIdx.x < 32)
        tdm_load_tile_bf16(z + (size_t)m0 * DD + (it + 1) * 32,
                           (unsigned)(size_t)(void*)As[buf ^ 1],
                           DD, NTOK, DD, 32, 128);
    }

    // compute on bank 'buf'
    const int rl = lane & 15;
    v16bf a[2], g[2], u[2];
#pragma unroll
    for (int s = 0; s < 2; ++s) {
      a[s] = frag_from_lds(As[buf],  (2 * wm + s) * 16 + rl, SA, lane);
      g[s] = frag_from_lds(Bgs[buf], (2 * wn + s) * 16 + rl, SB, lane);
      u[s] = frag_from_lds(Bus[buf], (2 * wn + s) * 16 + rl, SB, lane);
    }
#pragma unroll
    for (int sm = 0; sm < 2; ++sm)
#pragma unroll
      for (int sn = 0; sn < 2; ++sn) {
        accg[sm][sn] = wmma_bf16(a[sm], g[sn], accg[sm][sn]);
        accu[sm][sn] = wmma_bf16(a[sm], u[sn], accu[sm][sn]);
      }

    if (more) {
      store_b_lds(ng, Bgs[buf ^ 1]);
      store_b_lds(nu, Bus[buf ^ 1]);
      if (threadIdx.x < 32) __builtin_amdgcn_s_wait_tensorcnt(0);
    }
  }

  // Epilogue. C/D layout: VGPR r -> M = r + 8*(lane/16), N = lane%16.
  const int nl = lane & 15;
  const int mh = (lane >> 4) * 8;
#pragma unroll
  for (int sm = 0; sm < 2; ++sm)
#pragma unroll
    for (int sn = 0; sn < 2; ++sn) {
      const int n = n0 + (2 * wn + sn) * 16 + nl;
      const float bgv = bg[n], buv = bu[n];
#pragma unroll
      for (int r = 0; r < 8; ++r) {
        const int m = m0 + (2 * wm + sm) * 16 + mh + r;
        const float gv = accg[sm][sn][r] + bgv;
        const float uv = accu[sm][sn][r] + buv;
        const float hv = (gv / (1.f + __expf(-gv))) * uv;  // silu(g)*u
        hbuf[(size_t)m * HH + n] = (__bf16)hv;
      }
    }
}

// GEMM2: out = xr + h@Wd + bd. M=8192, N=1024, K=4096. Same pipelined tiling.
__global__ void __launch_bounds__(256)
gemm_down_kernel(const __bf16* __restrict__ hbuf, const __bf16* __restrict__ wd,
                 const float* __restrict__ bd, const float* __restrict__ xr,
                 float* __restrict__ out) {
  const int m0 = blockIdx.x * 128;
  const int n0 = blockIdx.y * 64;
  const int lane = threadIdx.x & 31;
  const int wm = (threadIdx.x >> 5) >> 1;
  const int wn = (threadIdx.x >> 5) & 1;

  __shared__ unsigned short As[2][128 * SA];
  __shared__ unsigned short Bs[2][64 * SB];

  v8f acc[2][2] = {};

  constexpr int NIT = HH / 32;

  if (threadIdx.x < 32)
    tdm_load_tile_bf16(hbuf + (size_t)m0 * HH, (unsigned)(size_t)(void*)As[0],
                       HH, NTOK, HH, 32, 128);
  store_b_lds(load_b_regs(wd, DD, 0, n0), Bs[0]);
  if (threadIdx.x < 32) __builtin_amdgcn_s_wait_tensorcnt(0);

  for (int it = 0; it < NIT; ++it) {
    const int buf = it & 1;
    __syncthreads();

    u32x4 nb;
    const bool more = (it + 1) < NIT;
    if (more) {
      nb = load_b_regs(wd, DD, (it + 1) * 32, n0);
      if (threadIdx.x < 32)
        tdm_load_tile_bf16(hbuf + (size_t)m0 * HH + (it + 1) * 32,
                           (unsigned)(size_t)(void*)As[buf ^ 1],
                           HH, NTOK, HH, 32, 128);
    }

    const int rl = lane & 15;
    v16bf a[2], b[2];
#pragma unroll
    for (int s = 0; s < 2; ++s) {
      a[s] = frag_from_lds(As[buf], (2 * wm + s) * 16 + rl, SA, lane);
      b[s] = frag_from_lds(Bs[buf], (2 * wn + s) * 16 + rl, SB, lane);
    }
#pragma unroll
    for (int sm = 0; sm < 2; ++sm)
#pragma unroll
      for (int sn = 0; sn < 2; ++sn)
        acc[sm][sn] = wmma_bf16(a[sm], b[sn], acc[sm][sn]);

    if (more) {
      store_b_lds(nb, Bs[buf ^ 1]);
      if (threadIdx.x < 32) __builtin_amdgcn_s_wait_tensorcnt(0);
    }
  }

  const int nl = lane & 15;
  const int mh = (lane >> 4) * 8;
#pragma unroll
  for (int sm = 0; sm < 2; ++sm)
#pragma unroll
    for (int sn = 0; sn < 2; ++sn) {
      const int n = n0 + (2 * wn + sn) * 16 + nl;
      const float bdv = bd[n];
#pragma unroll
      for (int r = 0; r < 8; ++r) {
        const int m = m0 + (2 * wm + sm) * 16 + mh + r;
        out[(size_t)m * DD + n] = xr[(size_t)m * DD + n] + acc[sm][sn][r] + bdv;
      }
    }
}

// ---------------- host launcher ----------------
extern "C" void kernel_launch(void* const* d_in, const int* in_sizes, int n_in,
                              void* d_out, int out_size, void* d_ws, size_t ws_size,
                              hipStream_t stream) {
  (void)in_sizes; (void)n_in; (void)out_size; (void)ws_size;

  const float* x       = (const float*)d_in[0];
  const float* norm_w  = (const float*)d_in[1];
  const float* conv_w  = (const float*)d_in[2];
  const float* conv_b  = (const float*)d_in[3];
  const float* a       = (const float*)d_in[4];
  const float* bvec    = (const float*)d_in[5];
  const float* mlp_w   = (const float*)d_in[6];
  const float* w_gate  = (const float*)d_in[7];
  const float* b_gate  = (const float*)d_in[8];
  const float* w_up    = (const float*)d_in[9];
  const float* b_up    = (const float*)d_in[10];
  const float* w_down  = (const float*)d_in[11];
  const float* b_down  = (const float*)d_in[12];
  float* out           = (float*)d_out;

  char* ws = (char*)d_ws;
  size_t off = 0;
  float* y_ws    = (float*)(ws + off); off += (size_t)NTOK * DD * sizeof(float);
  float* xr_ws   = (float*)(ws + off); off += (size_t)NTOK * DD * sizeof(float);
  __bf16* z_ws   = (__bf16*)(ws + off); off += (size_t)NTOK * DD * sizeof(__bf16);
  __bf16* wg_ws  = (__bf16*)(ws + off); off += (size_t)DD * HH * sizeof(__bf16);
  __bf16* wu_ws  = (__bf16*)(ws + off); off += (size_t)DD * HH * sizeof(__bf16);
  __bf16* wd_ws  = (__bf16*)(ws + off); off += (size_t)HH * DD * sizeof(__bf16);
  __bf16* h_ws   = (__bf16*)(ws + off); off += (size_t)NTOK * HH * sizeof(__bf16);
  float* carry   = (float*)(ws + off); off += (size_t)BB * DD * NC * sizeof(float);
  float* prefix  = (float*)(ws + off); off += (size_t)BB * DD * NC * sizeof(float);

  {
    const int n = DD * HH;
    f32_to_bf16_kernel<<<n / 256, 256, 0, stream>>>(w_gate, wg_ws, n);
    f32_to_bf16_kernel<<<n / 256, 256, 0, stream>>>(w_up,   wu_ws, n);
    f32_to_bf16_kernel<<<n / 256, 256, 0, stream>>>(w_down, wd_ws, n);
  }
  rmsnorm_kernel<false><<<NTOK, 256, 0, stream>>>(x, norm_w, (void*)y_ws);
  conv_ssm_local_kernel<<<(BB * DD * NC) / 256, 256, 0, stream>>>(x, y_ws, conv_w, conv_b,
                                                                  a, bvec, xr_ws, carry);
  scan_carries_kernel<<<(BB * DD) / 256, 256, 0, stream>>>(a, carry, prefix);
  ssm_fixup_kernel<<<(BB * DD * NC) / 256, 256, 0, stream>>>(a, prefix, xr_ws);
  rmsnorm_kernel<true><<<NTOK, 256, 0, stream>>>(xr_ws, mlp_w, (void*)z_ws);
  gemm_gate_up_kernel<<<dim3(NTOK / 128, HH / 64), 256, 0, stream>>>(z_ws, wg_ws, wu_ws,
                                                                     b_gate, b_up, h_ws);
  gemm_down_kernel<<<dim3(NTOK / 128, DD / 64), 256, 0, stream>>>(h_ws, wd_ws, b_down,
                                                                  xr_ws, out);
}
